// RoPEAttention_12214886990797
// MI455X (gfx1250) — compile-verified, matrix operand order checked by reading, not run on hardware
//
#include <hip/hip_runtime.h>
#include <hip/hip_bf16.h>
#include <math.h>

// Problem constants (match reference)
#define B_N     2
#define S_N     2048
#define HID_N   1024
#define HEADS_N 16
#define HD_N    64
#define M_N     (B_N * S_N)      // 4096 rows of x

typedef __attribute__((ext_vector_type(16))) __bf16 v16bf;
typedef __attribute__((ext_vector_type(8)))  __bf16 v8bf;
typedef __attribute__((ext_vector_type(8)))  float  v8f;
typedef __attribute__((ext_vector_type(4)))  unsigned int v4ui;
typedef __attribute__((ext_vector_type(8)))  int  v8si;
typedef __attribute__((ext_vector_type(4)))  int  v4si;

// TDM availability (device pass only; host pass parses the fallback path)
#if defined(__HIP_DEVICE_COMPILE__) && defined(__has_builtin)
#  if __has_builtin(__builtin_amdgcn_tensor_load_to_lds) && \
      __has_builtin(__builtin_amdgcn_s_wait_tensorcnt)
#    define HAVE_TDM 1
#  else
#    define HAVE_TDM 0
#  endif
#else
#  define HAVE_TDM 0
#endif

// ---------------------------------------------------------------------------
// WMMA helpers (CDNA5 wave32, v_wmma_f32_16x16x32_bf16)
// ---------------------------------------------------------------------------
__device__ __forceinline__ v8f wmma_bf16(v16bf a, v16bf b, v8f c) {
  return __builtin_amdgcn_wmma_f32_16x16x32_bf16(false, a, false, b,
                                                 (short)0, c, false, false);
}

// Load a 16x32 bf16 fragment from a row-major (rows x ld) array.
// ISA layout for 16-bit A 16x32: lanes 0-15 row M=lane, K=0..7 & 16..23;
// lanes 16-31 row M=lane-16, K=8..15 & 24..31.  (B from an NxK array is
// symmetric.)
__device__ __forceinline__ v16bf load_frag(const __bf16* p, int row0, int ld,
                                           int k0, int lane) {
  int r    = row0 + (lane & 15);
  int koff = k0 + ((lane >> 4) << 3);
  const __bf16* q = p + (size_t)r * ld + koff;
  v8bf lo = *(const v8bf*)(q);
  v8bf hi = *(const v8bf*)(q + 16);
  v16bf f;
#pragma unroll
  for (int i = 0; i < 8; ++i) { f[i] = lo[i]; f[8 + i] = hi[i]; }
  return f;
}

// Flat LDS pointer -> LDS byte offset (ISA 10.2: low 32 bits of the address)
__device__ __forceinline__ unsigned lds_off(const void* p) {
  return (unsigned)(size_t)p;
}

#if HAVE_TDM
// Issue one TDM 2D tile load: global (row-major, stride0 elems of 2B) -> LDS
// packed tile_w x tile_h.  D# layout per CDNA5 ISA.
__device__ __forceinline__ void tdm_load_2d(unsigned lds_addr, const void* gptr,
                                            unsigned stride0_elems,
                                            unsigned tile_w, unsigned tile_h) {
  unsigned long long ga = (unsigned long long)(size_t)gptr;
  v4ui g0;
  g0[0] = 1u;                                   // count=1, user mode
  g0[1] = lds_addr;                             // lds byte address
  g0[2] = (unsigned)(ga & 0xffffffffu);         // global_addr[31:0]
  g0[3] = (unsigned)((ga >> 32) & 0x01ffffffu)  // global_addr[56:32]
          | 0x80000000u;                        // type = 2 ("image")
  v8si g1;
  g1[0] = (int)(1u << 16);                      // data_size=1 -> 2 bytes
  g1[1] = (int)((tile_w & 0xffffu) << 16);      // tensor_dim0 = tile_w
  g1[2] = (int)(((tile_w >> 16) & 0xffffu) |
                ((tile_h & 0xffffu) << 16));    // tensor_dim1 = tile_h
  g1[3] = (int)(((tile_h >> 16) & 0xffffu) |
                ((tile_w & 0xffffu) << 16));    // tile_dim0 = tile_w
  g1[4] = (int)(tile_h & 0xffffu);              // tile_dim1 = tile_h
  g1[5] = (int)stride0_elems;                   // tensor_dim0_stride[31:0]
  g1[6] = 0;
  g1[7] = 0;
  v4si gz = {0, 0, 0, 0};
#if __clang_major__ >= 23
  v8si gz8 = {0, 0, 0, 0, 0, 0, 0, 0};
  __builtin_amdgcn_tensor_load_to_lds(g0, g1, gz, gz, gz8, 0);
#else
  __builtin_amdgcn_tensor_load_to_lds(g0, g1, gz, gz, 0);
#endif
}
#endif

// ---------------------------------------------------------------------------
// fp32 -> bf16 cast
// ---------------------------------------------------------------------------
__global__ void cast_f32_bf16(const float* __restrict__ src,
                              __bf16* __restrict__ dst, size_t n) {
  size_t i = (size_t)blockIdx.x * blockDim.x + threadIdx.x;
  if (i < n) dst[i] = (__bf16)src[i];
}

// ---------------------------------------------------------------------------
// C(MxN) fp32 = A(MxK) bf16 @ Bt(NxK)^T bf16   (x @ W^T with W row-major)
// 256 threads = 8 waves; tile 128(M) x 128(N); K-step 64, double-buffered LDS
// filled by the Tensor Data Mover (fallback: cooperative b128 copies).
// ---------------------------------------------------------------------------
#define TILE_M 128
#define TILE_N 128
#define TILE_K 64

__device__ __forceinline__ void stage_tiles(const __bf16* A, const __bf16* Bt,
                                            int K, __bf16* abuf, __bf16* bbuf,
                                            int m0, int n0, int kk, int tid) {
#if HAVE_TDM
  if (tid < 32) {   // wave 0 drives the TDM; EXEC is ignored by tensor ops
    tdm_load_2d(lds_off(abuf), A + (size_t)m0 * K + kk, (unsigned)K,
                TILE_K, TILE_M);
    tdm_load_2d(lds_off(bbuf), Bt + (size_t)n0 * K + kk, (unsigned)K,
                TILE_K, TILE_N);
  }
#else
  // Cooperative copy: rows of TILE_K bf16 = 128B = 8 chunks of 16B
  for (int c = tid; c < (TILE_M * TILE_K) / 8; c += 256) {
    int row = c >> 3, ch = c & 7;
    *(v8bf*)(abuf + row * TILE_K + ch * 8) =
        *(const v8bf*)(A + (size_t)(m0 + row) * K + kk + ch * 8);
  }
  for (int c = tid; c < (TILE_N * TILE_K) / 8; c += 256) {
    int row = c >> 3, ch = c & 7;
    *(v8bf*)(bbuf + row * TILE_K + ch * 8) =
        *(const v8bf*)(Bt + (size_t)(n0 + row) * K + kk + ch * 8);
  }
#endif
}

__global__ void __launch_bounds__(256)
gemm_bf16_nt(const __bf16* __restrict__ A, const __bf16* __restrict__ Bt,
             float* __restrict__ C, int M, int N, int K) {
  __shared__ __bf16 Ab[2][TILE_M * TILE_K];   // 2 x 16 KB
  __shared__ __bf16 Bb[2][TILE_N * TILE_K];   // 2 x 16 KB
  int tid = threadIdx.x;
  int lane = tid & 31;
  int wave = tid >> 5;
  int m0 = blockIdx.x * TILE_M;
  int n0 = blockIdx.y * TILE_N;

  v8f acc[8] = {v8f{}, v8f{}, v8f{}, v8f{}, v8f{}, v8f{}, v8f{}, v8f{}};

  stage_tiles(A, Bt, K, Ab[0], Bb[0], m0, n0, 0, tid);

  for (int kk = 0; kk < K; kk += TILE_K) {
    int cur = (kk / TILE_K) & 1;
#if HAVE_TDM
    if (tid < 32) __builtin_amdgcn_s_wait_tensorcnt(0);
#endif
    __syncthreads();  // staged data visible to whole WGP
    if (kk + TILE_K < K)
      stage_tiles(A, Bt, K, Ab[cur ^ 1], Bb[cur ^ 1], m0, n0, kk + TILE_K, tid);

    // Inner product: B fragments register-double-buffered so ds_loads lead
    // their consuming WMMA by two subtiles (coarser s_wait_dscnt).
#pragma unroll
    for (int ks = 0; ks < 2; ++ks) {
      v16bf a  = load_frag(Ab[cur], wave * 16, TILE_K, ks * 32, lane);
      v16bf b0 = load_frag(Bb[cur], 0,  TILE_K, ks * 32, lane);
      v16bf b1 = load_frag(Bb[cur], 16, TILE_K, ks * 32, lane);
#pragma unroll
      for (int n = 0; n < 8; ++n) {
        v16bf bn = (n & 1) ? b1 : b0;
        v16bf bnew = bn;
        if (n + 2 < 8)
          bnew = load_frag(Bb[cur], (n + 2) * 16, TILE_K, ks * 32, lane);
        acc[n] = wmma_bf16(a, bn, acc[n]);
        if (n & 1) b1 = bnew; else b0 = bnew;
      }
    }
    __syncthreads();  // done reading 'cur' before it is restaged
  }

  int rbase = m0 + wave * 16 + ((lane >> 4) << 3);
  int cbase = n0 + (lane & 15);
#pragma unroll
  for (int n = 0; n < 8; ++n)
#pragma unroll
    for (int v = 0; v < 8; ++v)
      C[(size_t)(rbase + v) * N + cbase + n * 16] = acc[n][v];
}

// ---------------------------------------------------------------------------
// Per-head RMSNorm + RoPE + optional 1/sqrt(HD) scale, fp32 -> bf16.
// One wave32 per head-row of 64 elems (32 rotary pairs). 4 rows per block.
// ---------------------------------------------------------------------------
__global__ void __launch_bounds__(128)
rmsnorm_rope_kernel(const float* __restrict__ in, const float* __restrict__ w,
                    const float* __restrict__ cosT, const float* __restrict__ sinT,
                    __bf16* __restrict__ out, float scale, int nrows) {
  int lane = threadIdx.x & 31;
  int row  = blockIdx.x * 4 + (threadIdx.x >> 5);
  if (row >= nrows) return;
  int s = (row / HEADS_N) % S_N;
  size_t base = (size_t)row * HD_N;

  float xr = in[base + 2 * lane];
  float xi = in[base + 2 * lane + 1];

  float ss = xr * xr + xi * xi;
#pragma unroll
  for (int mask = 1; mask < 32; mask <<= 1)
    ss += __shfl_xor(ss, mask, 32);
  float r = rsqrtf(ss * (1.0f / (float)HD_N) + 1e-6f);

  float nr = xr * r * w[2 * lane];
  float ni = xi * r * w[2 * lane + 1];
  float c  = cosT[s * (HD_N / 2) + lane];
  float sn = sinT[s * (HD_N / 2) + lane];
  float orr = (nr * c - ni * sn) * scale;
  float oii = (nr * sn + ni * c) * scale;

  out[base + 2 * lane]     = (__bf16)orr;
  out[base + 2 * lane + 1] = (__bf16)oii;
}

// ---------------------------------------------------------------------------
// Flash attention: one block per (64-query tile, b*h). 128 threads = 4 waves.
// Q,K,V in bf16 laid out (B,S,HEADS,HD). Online softmax over key blocks of 64.
// K block staged to LDS by the TDM (shared by all 4 waves); V staged
// transposed by cooperative copy; P round-trips LDS for the layout change.
// ---------------------------------------------------------------------------
__global__ void __launch_bounds__(128)
attn_kernel(const __bf16* __restrict__ Q, const __bf16* __restrict__ Km,
            const __bf16* __restrict__ V, float* __restrict__ O) {
  __shared__ __bf16 kt[64 * HD_N];        // K tile: [key][hd]    (8 KB)
  __shared__ __bf16 vt[HD_N * 64];        // V^T tile: [hd][key]  (8 KB)
  __shared__ __bf16 pbuf[4][16 * 64];     // per-wave P tile      (8 KB)

  int tid = threadIdx.x;
  int lane = tid & 31;
  int wave = tid >> 5;
  int q0 = blockIdx.x * 64;
  int bh = blockIdx.y;
  int b = bh / HEADS_N, h = bh % HEADS_N;

  const __bf16* qbase = Q + (size_t)b * S_N * HID_N + h * HD_N;
  const __bf16* kbase = Km + (size_t)b * S_N * HID_N + h * HD_N;
  const __bf16* vbase = V + (size_t)b * S_N * HID_N + h * HD_N;

  int mq = q0 + wave * 16;  // this wave's query-row base

  // Q strip is invariant over key blocks: load the two k-step fragments once.
  v16bf aq0 = load_frag(qbase, mq, HID_N, 0, lane);
  v16bf aq1 = load_frag(qbase, mq, HID_N, 32, lane);

  v8f acc[4] = {v8f{}, v8f{}, v8f{}, v8f{}};
  float mrow[8], lrow[8];
#pragma unroll
  for (int v = 0; v < 8; ++v) { mrow[v] = -3.0e38f; lrow[v] = 0.0f; }

  for (int k0 = 0; k0 < S_N; k0 += 64) {
    __syncthreads();  // previous readers of kt/vt done
#if HAVE_TDM
    // Async K tile DMA overlaps the cooperative V transpose below.
    if (tid < 32)
      tdm_load_2d(lds_off(kt), kbase + (size_t)k0 * HID_N, HID_N, HD_N, 64);
#endif
    // Stage V block transposed: vt[hd][key]
    for (int e = tid; e < 64 * HD_N; e += 128) {
      int key = e >> 6;
      int d   = e & 63;
      vt[d * 64 + key] = vbase[(size_t)(k0 + key) * HID_N + d];
    }
#if HAVE_TDM
    if (tid < 32) __builtin_amdgcn_s_wait_tensorcnt(0);
#else
    for (int c = tid; c < (64 * HD_N) / 8; c += 128) {
      int row = c >> 3, ch = c & 7;
      *(v8bf*)(kt + row * HD_N + ch * 8) =
          *(const v8bf*)(kbase + (size_t)(k0 + row) * HID_N + ch * 8);
    }
#endif
    if (k0 + 64 < S_N)
      __builtin_prefetch(vbase + (size_t)(k0 + 64) * HID_N, 0, 0);
    __syncthreads();

    // S strip = Q(16x64) @ K_block^T(64x64): 4 N-subtiles, 2 k-steps, LDS-fed
    v8f st[4];
#pragma unroll
    for (int n = 0; n < 4; ++n) {
      v8f c = v8f{};
      v16bf bk0 = load_frag(kt, n * 16, HD_N, 0, lane);
      c = wmma_bf16(aq0, bk0, c);
      v16bf bk1 = load_frag(kt, n * 16, HD_N, 32, lane);
      c = wmma_bf16(aq1, bk1, c);
      st[n] = c;
    }

    // Online softmax. Row of st[n][v]: (lane>=16 ? 8:0)+v; col: n*16+(lane&15).
    // shfl_xor masks 1,2,4,8 reduce within 16-lane halves == within a row.
#pragma unroll
    for (int v = 0; v < 8; ++v) {
      float rm = st[0][v];
#pragma unroll
      for (int n = 1; n < 4; ++n) rm = fmaxf(rm, st[n][v]);
#pragma unroll
      for (int mask = 1; mask < 16; mask <<= 1)
        rm = fmaxf(rm, __shfl_xor(rm, mask, 32));
      float mn   = fmaxf(mrow[v], rm);
      float corr = __expf(mrow[v] - mn);
      float rs = 0.0f;
#pragma unroll
      for (int n = 0; n < 4; ++n) {
        float p = __expf(st[n][v] - mn);
        st[n][v] = p;
        rs += p;
      }
#pragma unroll
      for (int mask = 1; mask < 16; mask <<= 1)
        rs += __shfl_xor(rs, mask, 32);
      lrow[v] = lrow[v] * corr + rs;
      mrow[v] = mn;
#pragma unroll
      for (int n = 0; n < 4; ++n) acc[n][v] *= corr;
    }

    // Spill P (C-layout) to LDS as a row-major 16x64 bf16 tile (A-layout feed)
    int rloc = ((lane >> 4) << 3);
#pragma unroll
    for (int n = 0; n < 4; ++n)
#pragma unroll
      for (int v = 0; v < 8; ++v)
        pbuf[wave][(rloc + v) * 64 + n * 16 + (lane & 15)] = (__bf16)st[n][v];
    __syncthreads();

    // O += P(16x64) @ V_block(64x64): A from pbuf, B from vt (both LDS)
#pragma unroll
    for (int ks = 0; ks < 2; ++ks) {
      v16bf ap = load_frag(&pbuf[wave][0], 0, 64, ks * 32, lane);
#pragma unroll
      for (int n = 0; n < 4; ++n) {
        v16bf bv = load_frag(vt, n * 16, 64, ks * 32, lane);
        acc[n] = wmma_bf16(ap, bv, acc[n]);
      }
    }
  }

  // Epilogue: normalize and store fp32 (B,S,HEADS,HD)
  int rbase = mq + ((lane >> 4) << 3);
#pragma unroll
  for (int n = 0; n < 4; ++n)
#pragma unroll
    for (int v = 0; v < 8; ++v)
      O[(size_t)(b * S_N + rbase + v) * HID_N + h * HD_N + n * 16 + (lane & 15)] =
          acc[n][v] / lrow[v];
}

// ---------------------------------------------------------------------------
// Host-side orchestration
// ---------------------------------------------------------------------------
extern "C" void kernel_launch(void* const* d_in, const int* in_sizes, int n_in,
                              void* d_out, int out_size, void* d_ws, size_t ws_size,
                              hipStream_t stream) {
  (void)in_sizes; (void)n_in; (void)out_size; (void)ws_size;

  const float* x    = (const float*)d_in[0];
  const float* Wq   = (const float*)d_in[1];
  const float* Wk   = (const float*)d_in[2];
  const float* Wv   = (const float*)d_in[3];
  const float* Wo   = (const float*)d_in[4];
  const float* qn_w = (const float*)d_in[5];
  const float* kn_w = (const float*)d_in[6];
  const float* cosT = (const float*)d_in[7];
  const float* sinT = (const float*)d_in[8];

  char* ws = (char*)d_ws;
  const size_t MB = (size_t)1 << 20;

  __bf16* xb  = (__bf16*)(ws + 0 * MB);   // 8 MB   x in bf16
  __bf16* wqb = (__bf16*)(ws + 8 * MB);   // 2 MB each
  __bf16* wkb = (__bf16*)(ws + 10 * MB);
  __bf16* wvb = (__bf16*)(ws + 12 * MB);
  __bf16* wob = (__bf16*)(ws + 14 * MB);
  float*  qf  = (float*)(ws + 16 * MB);   // 16 MB fp32 Q   (reused as attnf)
  float*  kf  = (float*)(ws + 32 * MB);   // 16 MB fp32 K   (reused as attnb)
  float*  vf  = (float*)(ws + 48 * MB);   // 16 MB fp32 V
  __bf16* qb  = (__bf16*)(ws + 64 * MB);  // 8 MB bf16 Q (normed+roped+scaled)
  __bf16* kb  = (__bf16*)(ws + 72 * MB);  // 8 MB bf16 K (normed+roped)
  __bf16* vb  = (__bf16*)(ws + 80 * MB);  // 8 MB bf16 V
  float*  attnf = qf;                      // reuse: fp32 attention output
  __bf16* attnb = (__bf16*)kf;             // reuse: bf16 attention output

  const size_t nx = (size_t)M_N * HID_N;       // 4 Mi elems
  const size_t nw = (size_t)HID_N * HID_N;     // 1 Mi elems

  // 1) casts to bf16
  cast_f32_bf16<<<(nx + 255) / 256, 256, 0, stream>>>(x, xb, nx);
  cast_f32_bf16<<<(nw + 255) / 256, 256, 0, stream>>>(Wq, wqb, nw);
  cast_f32_bf16<<<(nw + 255) / 256, 256, 0, stream>>>(Wk, wkb, nw);
  cast_f32_bf16<<<(nw + 255) / 256, 256, 0, stream>>>(Wv, wvb, nw);
  cast_f32_bf16<<<(nw + 255) / 256, 256, 0, stream>>>(Wo, wob, nw);

  // 2) QKV projections (TDM-fed WMMA GEMM, 128x128 tiles)
  dim3 ggrid(M_N / TILE_M, HID_N / TILE_N);
  gemm_bf16_nt<<<ggrid, 256, 0, stream>>>(xb, wqb, qf, M_N, HID_N, HID_N);
  gemm_bf16_nt<<<ggrid, 256, 0, stream>>>(xb, wkb, kf, M_N, HID_N, HID_N);
  gemm_bf16_nt<<<ggrid, 256, 0, stream>>>(xb, wvb, vf, M_N, HID_N, HID_N);

  // 3) RMSNorm + RoPE (+ 1/sqrt(HD) for Q), cast to bf16
  const int nrows = M_N * HEADS_N;  // 65536 head-rows
  rmsnorm_rope_kernel<<<nrows / 4, 128, 0, stream>>>(qf, qn_w, cosT, sinT, qb,
                                                     0.125f, nrows);
  rmsnorm_rope_kernel<<<nrows / 4, 128, 0, stream>>>(kf, kn_w, cosT, sinT, kb,
                                                     1.0f, nrows);
  cast_f32_bf16<<<(nx + 255) / 256, 256, 0, stream>>>(vf, vb, nx);

  // 4) flash attention per (query-tile, b*h)
  dim3 agrid(S_N / 64, B_N * HEADS_N);
  attn_kernel<<<agrid, 128, 0, stream>>>(qb, kb, vb, attnf);

  // 5) output projection
  cast_f32_bf16<<<(nx + 255) / 256, 256, 0, stream>>>(attnf, attnb, nx);
  gemm_bf16_nt<<<ggrid, 256, 0, stream>>>(attnb, wob, (float*)d_out,
                                          M_N, HID_N, HID_N);
}
